// MultiHeadAttention_44281112822190
// MI455X (gfx1250) — matrix operand order repaired and checked by reading, not compiled, not run on hardware
//
#include <hip/hip_runtime.h>
#include <hip/hip_bf16.h>

typedef __attribute__((ext_vector_type(16))) __bf16    v16bf;
typedef __attribute__((ext_vector_type(8)))  float     v8f;
typedef __attribute__((ext_vector_type(4)))  unsigned  u32x4;
typedef __attribute__((ext_vector_type(8)))  unsigned  u32x8;

#define B_  4
#define S_  2048
#define E_  2048
#define H_  16
#define D_  128
#define D2_ 64
#define M_  (B_ * S_)   // 8192

#define LDA_ 40         // padded LDS row stride (elements): 64B data + 16B pad

__device__ __forceinline__ __bf16 f2bf(float f) { return (__bf16)f; }

// ---------------------------------------------------------------- f32 -> bf16
__global__ __launch_bounds__(256)
void cvt_f32_bf16_kernel(const float* __restrict__ src, __bf16* __restrict__ dst, int n) {
    int i = (blockIdx.x * 256 + threadIdx.x) * 4;
    if (i >= n) return;
    float4 v = *(const float4*)(src + i);
    __bf16 o[4] = { f2bf(v.x), f2bf(v.y), f2bf(v.z), f2bf(v.w) };
    *(uint2*)(dst + i) = *(const uint2*)o;
}

// ---------------------------------------------------- async global -> LDS copy
// One b128 per lane, tracked by ASYNCcnt.
__device__ __forceinline__ void async_b128(unsigned lds_off, const __bf16* g) {
    asm volatile("global_load_async_to_lds_b128 %0, %1, off"
                 :: "v"(lds_off), "v"(g) : "memory");
}

// ---------------------------------------------------- TDM tile load (2D)
// One tensor_load_to_lds moves a 128x32 bf16 tile (row stride Kdim elements)
// into LDS at lds_off, inserting 16B of padding after every 64B row
// (pad_interval=3 -> 16 DWORDs, pad_amount=3 -> 4 DWORDs) to match LDA_=40.
// Tracked by TENSORcnt. D# per CDNA5 ISA.
__device__ __forceinline__ void tdm_load_tile(const __bf16* gbase, unsigned lds_off, int Kdim) {
    unsigned long long ga = (unsigned long long)(size_t)gbase;
    u32x4 g0;
    g0[0] = 1u;                                    // count=1 (valid), user mode
    g0[1] = lds_off;                               // lds_addr (bytes)
    g0[2] = (unsigned)ga;                          // global_addr[31:0]
    g0[3] = (unsigned)(ga >> 32) | (2u << 30);     // global_addr[56:32] | type=2
    unsigned td0 = (unsigned)Kdim;                 // tensor_dim0 (elements)
    unsigned td1 = 1u << 20;                       // tensor_dim1 (never OOB)
    u32x8 g1;
    g1[0] = (1u << 16)                             // data_size = 1 -> 2 bytes
          | (1u << 20)                             // pad_enable
          | (3u << 22)                             // pad_interval: 16 DWORDs
          | (3u << 25);                            // pad_amount:   4 DWORDs
    g1[1] = (td0 & 0xFFFFu) << 16;                 // tensor_dim0[15:0]
    g1[2] = (td0 >> 16) | ((td1 & 0xFFFFu) << 16); // tensor_dim0[31:16] | dim1 lo
    g1[3] = (td1 >> 16) | (32u << 16);             // dim1 hi | tile_dim0 = 32
    g1[4] = 128u;                                  // tile_dim1 = 128, tile_dim2 = 0
    g1[5] = (unsigned)Kdim;                        // tensor_dim0_stride[31:0]
    g1[6] = 0u;                                    // stride0 hi, stride1 lo
    g1[7] = 0u;                                    // stride1 hi
    asm volatile("tensor_load_to_lds %0, %1" :: "s"(g0), "s"(g1) : "memory");
}

// ------------------------------------------------- WMMA fragment loaders
// A-matrix 16x32 bf16 (row m = lane&15): per lane two 8-elem chunks at
// k = half*8 and k = 16 + half*8   (ISA 7.12.2, 16-bit A layout)
__device__ __forceinline__ v16bf frag_a(const __bf16* __restrict__ p0, int ld, int lane) {
    int lo = lane & 15, hi = lane >> 4;
    const __bf16* p = p0 + (size_t)lo * ld + hi * 8;
    union { v16bf v; uint4 q[2]; } t;
    t.q[0] = *(const uint4*)(p);
    t.q[1] = *(const uint4*)(p + 16);
    return t.v;
}
// B-matrix 32x16 bf16 (col n = lane&15): per lane two 8-elem chunks at
// k = half*8 and k = 16 + half*8 (same K order as frag_a).
__device__ __forceinline__ v16bf frag_b(const __bf16* __restrict__ p0, int ld, int lane) {
    int lo = lane & 15, hi = lane >> 4;
    const __bf16* p = p0 + (size_t)lo * ld + hi * 8;
    union { v16bf v; uint4 q[2]; } t;
    t.q[0] = *(const uint4*)(p);
    t.q[1] = *(const uint4*)(p + 16);
    return t.v;
}

#define WMMA_BF16(a, b, c) \
    __builtin_amdgcn_wmma_f32_16x16x32_bf16(false, (a), false, (b), (short)0, (c), false, false)

// ---------------------------------------------------------------- GEMM
// out[m,n] = sum_k A[m,k] * W[n,k] + bias[n]
// Block: 256 threads = 8 waves, block tile 128x128, K-step 32, double-buffered.
// A tiles staged by the Tensor Data Mover (wave 0, TENSORcnt); B tiles staged
// by per-lane async b128 copies (all waves, ASYNCcnt). Waves 4(M) x 2(N),
// wave tile 32x64 = 2x4 WMMA tiles (8 wmma per K-step).
template <bool OUT_BF16>
__global__ __launch_bounds__(256)
void gemm_bf16_kernel(const __bf16* __restrict__ A, const __bf16* __restrict__ W,
                      const float* __restrict__ bias, void* __restrict__ out,
                      int Ndim, int Kdim) {
    alignas(16) __shared__ __bf16 shA[2][128 * LDA_];
    alignas(16) __shared__ __bf16 shB[2][128 * LDA_];

    const int lane  = threadIdx.x & 31;
    const int w     = threadIdx.x >> 5;
    const int wm    = w & 3;     // 0..3 over M
    const int wn    = w >> 2;    // 0..1 over N
    const int m_blk = blockIdx.x * 128;
    const int n_blk = blockIdx.y * 128;

    // B-tile copy: each thread moves 2 b128 chunks per tile.
    // chunk q (0..511): row = q>>2, col = (q&3)*8 elements
    const int q0 = threadIdx.x * 2;
    const int r0 = q0 >> 2,       c0 = (q0 & 3) * 8;
    const int r1 = (q0 + 1) >> 2, c1 = ((q0 + 1) & 3) * 8;
    const __bf16* gB0 = W + (size_t)(n_blk + r0) * Kdim + c0;
    const __bf16* gB1 = W + (size_t)(n_blk + r1) * Kdim + c1;
    unsigned lB0[2], lB1[2], lA[2];
    #pragma unroll
    for (int b = 0; b < 2; ++b) {
        lB0[b] = (unsigned)(size_t)&shB[b][r0 * LDA_ + c0];
        lB1[b] = (unsigned)(size_t)&shB[b][r1 * LDA_ + c1];
        lA[b]  = (unsigned)(size_t)&shA[b][0];
    }
    const __bf16* gA = A + (size_t)m_blk * Kdim;   // A tile origin (k advances)

    // prologue: stage tile 0 into buffer 0
    async_b128(lB0[0], gB0);
    async_b128(lB1[0], gB1);
    if (w == 0) tdm_load_tile(gA, lA[0], Kdim);

    v8f acc[2][4] = {};
    const int nk = Kdim / 32;
    for (int kt = 0; kt < nk; ++kt) {
        const int buf = kt & 1;
        if (kt + 1 < nk) {
            const int nb = (kt + 1) & 1;
            const int ko = (kt + 1) * 32;
            async_b128(lB0[nb], gB0 + ko);
            async_b128(lB1[nb], gB1 + ko);
            if (w == 0) {
                tdm_load_tile(gA + ko, lA[nb], Kdim);
                __builtin_amdgcn_s_wait_tensorcnt(1);   // A tile kt landed
            }
            asm volatile("s_wait_asynccnt 0x2" ::: "memory");   // B tile kt landed
        } else {
            if (w == 0) __builtin_amdgcn_s_wait_tensorcnt(0);
            asm volatile("s_wait_asynccnt 0x0" ::: "memory");
        }
        __syncthreads();   // tile kt visible to all waves

        v16bf a0 = frag_a(&shA[buf][(wm * 32) * LDA_],      LDA_, lane);
        v16bf a1 = frag_a(&shA[buf][(wm * 32 + 16) * LDA_], LDA_, lane);
        #pragma unroll
        for (int tj = 0; tj < 4; ++tj) {
            v16bf bfrag = frag_b(&shB[buf][(wn * 64 + tj * 16) * LDA_], LDA_, lane);
            acc[0][tj] = WMMA_BF16(a0, bfrag, acc[0][tj]);
            acc[1][tj] = WMMA_BF16(a1, bfrag, acc[1][tj]);
        }
        __syncthreads();   // all reads of buf done before it is refilled
    }

    const int lo = lane & 15, hi = lane >> 4;
    const int m0 = m_blk + wm * 32;
    const int n0 = n_blk + wn * 64;
    #pragma unroll
    for (int ti = 0; ti < 2; ++ti)
        #pragma unroll
        for (int tj = 0; tj < 4; ++tj) {
            int col = n0 + tj * 16 + lo;
            float bv = bias[col];
            #pragma unroll
            for (int i = 0; i < 8; ++i) {
                int row = m0 + ti * 16 + hi * 8 + i;
                float v = acc[ti][tj][i] + bv;
                if (OUT_BF16) ((__bf16*)out)[(size_t)row * Ndim + col] = f2bf(v);
                else          ((float*)out)[(size_t)row * Ndim + col] = v;
            }
        }
}

// ---------------------------------------------------------------- RoPE (Q,K in place)
__global__ __launch_bounds__(256)
void rope_kernel(__bf16* __restrict__ Q, __bf16* __restrict__ K,
                 const float* __restrict__ fc, const float* __restrict__ fs) {
    int idx = blockIdx.x * 256 + threadIdx.x;        // [0, M_*H_*D2_)
    int j   = idx & (D2_ - 1);
    int h   = (idx >> 6) & (H_ - 1);
    int bs  = idx >> 10;
    if (bs >= M_) return;
    int s = bs & (S_ - 1);
    float c = fc[s * D2_ + j], sn = fs[s * D2_ + j];
    size_t off = (size_t)bs * E_ + h * D_ + 2 * j;
    float xr = (float)Q[off], xi = (float)Q[off + 1];
    Q[off]     = f2bf(xr * c - xi * sn);
    Q[off + 1] = f2bf(xr * sn + xi * c);
    xr = (float)K[off]; xi = (float)K[off + 1];
    K[off]     = f2bf(xr * c - xi * sn);
    K[off + 1] = f2bf(xr * sn + xi * c);
}

// ---------------------------------------------------------------- attention
// One wave per (b,s): 16x16 head-vs-head scores via WMMA (K=128), softmax over
// the 16-lane column dim, att@V in f32 VALU, scatter into the reference's
// permuted (b,h,s,d)->(b,s',e') layout (flat index b*S*E + h*S*D + s*D + d).
__global__ __launch_bounds__(256)
void attn_kernel(const __bf16* __restrict__ Q, const __bf16* __restrict__ K,
                 const __bf16* __restrict__ V, __bf16* __restrict__ outbuf) {
    __shared__ float att_s[8][16][16];
    const int lane = threadIdx.x & 31;
    const int w    = threadIdx.x >> 5;
    const int bs   = blockIdx.x * 8 + w;
    const int b    = bs >> 11;           // / S_
    const int s    = bs & (S_ - 1);

    const __bf16* Qt = Q + (size_t)bs * E_;   // 16 x 128, ld = 128
    const __bf16* Kt = K + (size_t)bs * E_;

    v8f acc = {};
    #pragma unroll
    for (int k0 = 0; k0 < D_; k0 += 32) {
        v16bf a  = frag_a(Qt + k0, D_, lane);
        v16bf bb = frag_b(Kt + k0, D_, lane);
        acc = WMMA_BF16(a, bb, acc);
    }

    const float scale = 0.08838834764831845f;   // 1/sqrt(128)
    const int lo = lane & 15, hi = lane >> 4;
    #pragma unroll
    for (int i = 0; i < 8; ++i) {
        float x  = acc[i] * scale;
        float mx = x;
        #pragma unroll
        for (int off = 1; off < 16; off <<= 1) mx = fmaxf(mx, __shfl_xor(mx, off, 32));
        float e = __expf(x - mx);
        float sum = e;
        #pragma unroll
        for (int off = 1; off < 16; off <<= 1) sum += __shfl_xor(sum, off, 32);
        att_s[w][hi * 8 + i][lo] = e / sum;
    }
    __syncthreads();

    const int h  = lane >> 1;
    const int d0 = (lane & 1) * 64;
    float o[64];
    #pragma unroll
    for (int d = 0; d < 64; ++d) o[d] = 0.f;
    for (int g = 0; g < 16; ++g) {
        float a = att_s[w][h][g];
        const __bf16* vr = V + (size_t)bs * E_ + g * D_ + d0;
        #pragma unroll
        for (int d = 0; d < 64; ++d) o[d] += a * (float)vr[d];
    }
    __bf16* dst = outbuf + (size_t)b * S_ * E_ + (size_t)h * S_ * D_ + (size_t)s * D_ + d0;
    #pragma unroll
    for (int d = 0; d < 64; ++d) dst[d] = f2bf(o[d]);
}

// ---------------------------------------------------------------- launch
extern "C" void kernel_launch(void* const* d_in, const int* in_sizes, int n_in,
                              void* d_out, int out_size, void* d_ws, size_t ws_size,
                              hipStream_t stream) {
    const float* x  = (const float*)d_in[0];
    const float* fc = (const float*)d_in[1];
    const float* fs = (const float*)d_in[2];
    const float* wq = (const float*)d_in[3];
    const float* bq = (const float*)d_in[4];
    const float* wk = (const float*)d_in[5];
    const float* bk = (const float*)d_in[6];
    const float* wv = (const float*)d_in[7];
    const float* bv = (const float*)d_in[8];
    const float* wo = (const float*)d_in[9];
    const float* bo = (const float*)d_in[10];

    char* p = (char*)d_ws;
    const size_t szX = (size_t)M_ * E_ * sizeof(__bf16);   // 32 MB
    const size_t szW = (size_t)E_ * E_ * sizeof(__bf16);   //  8 MB
    __bf16* Xb  = (__bf16*)p; p += szX;
    __bf16* Wqb = (__bf16*)p; p += szW;
    __bf16* Wkb = (__bf16*)p; p += szW;
    __bf16* Wvb = (__bf16*)p; p += szW;
    __bf16* Wob = (__bf16*)p; p += szW;
    __bf16* Qb  = (__bf16*)p; p += szX;
    __bf16* Kb  = (__bf16*)p; p += szX;
    __bf16* Vb  = (__bf16*)p; p += szX;
    __bf16* Ab  = (__bf16*)p; p += szX;   // permuted attention output

    const int nX = M_ * E_;     // 16,777,216
    const int nW = E_ * E_;     //  4,194,304
    cvt_f32_bf16_kernel<<<nX / 4 / 256, 256, 0, stream>>>(x,  Xb,  nX);
    cvt_f32_bf16_kernel<<<nW / 4 / 256, 256, 0, stream>>>(wq, Wqb, nW);
    cvt_f32_bf16_kernel<<<nW / 4 / 256, 256, 0, stream>>>(wk, Wkb, nW);
    cvt_f32_bf16_kernel<<<nW / 4 / 256, 256, 0, stream>>>(wv, Wvb, nW);
    cvt_f32_bf16_kernel<<<nW / 4 / 256, 256, 0, stream>>>(wo, Wob, nW);

    dim3 gg(M_ / 128, E_ / 128);   // (64, 16)
    gemm_bf16_kernel<true><<<gg, 256, 0, stream>>>(Xb, Wqb, bq, Qb, E_, E_);
    gemm_bf16_kernel<true><<<gg, 256, 0, stream>>>(Xb, Wkb, bk, Kb, E_, E_);
    gemm_bf16_kernel<true><<<gg, 256, 0, stream>>>(Xb, Wvb, bv, Vb, E_, E_);

    rope_kernel<<<(M_ * H_ * D2_) / 256, 256, 0, stream>>>(Qb, Kb, fc, fs);

    attn_kernel<<<M_ / 8, 256, 0, stream>>>(Qb, Kb, Vb, Ab);

    gemm_bf16_kernel<false><<<gg, 256, 0, stream>>>(Ab, Wob, bo, d_out, E_, E_);
}
